// GSL_5334349382210
// MI455X (gfx1250) — matrix operand order repaired and checked by reading, not compiled, not run on hardware
//
#include <hip/hip_runtime.h>
#include <hip/hip_bf16.h>
#include <stdint.h>

#define Nn 8192
#define Ff 512
#define Hh 256
#define Ee 128
#define Cc 16

typedef __bf16   v16bf __attribute__((ext_vector_type(16)));
typedef float    v8f   __attribute__((ext_vector_type(8)));
typedef float    f4v   __attribute__((ext_vector_type(4)));
typedef unsigned u4v   __attribute__((ext_vector_type(4)));
typedef unsigned u2v   __attribute__((ext_vector_type(2)));

union FragBF { u4v q[2]; v16bf v; };

__device__ __forceinline__ unsigned short f2bf(float x) {
    unsigned u = __float_as_uint(x);
    unsigned r = u + 0x7FFFu + ((u >> 16) & 1u);   // round-to-nearest-even
    return (unsigned short)(r >> 16);
}
__device__ __forceinline__ unsigned pk2(float a, float b) {
    return (unsigned)f2bf(a) | ((unsigned)f2bf(b) << 16);
}
// order-preserving fp32 -> u32 key (ascending)
__device__ __forceinline__ unsigned ordkey(float f) {
    unsigned u = __float_as_uint(f);
    return (u & 0x80000000u) ? ~u : (u | 0x80000000u);
}
__device__ __forceinline__ float ordinv(unsigned k) {
    unsigned u = (k & 0x80000000u) ? (k ^ 0x80000000u) : ~k;
    return __uint_as_float(u);
}

// ---------------------------------------------------------------------------
// Generic bf16 WMMA GEMM: C[M x Ncols] = op(A) @ op(B), fp32 in memory,
// bf16 in LDS/WMMA, fp32 accumulate.
//   bt=1   : B is [Ncols x K] row-major (computes A @ B^T)  -- used for Z Z^T
//   normA=1: A element = (A[r][c] + (r==c)) * dinv[r] * dinv[c]  (fused sym_norm)
// Block tile 128x128, BK=32. 256 threads = 8 waves (wave32), wave grid 2(M)x4(N),
// each wave computes 4x2 tiles of v_wmma_f32_16x16x32_bf16.
// Requires: M % 128 == 0, K % 32 == 0 (true for all call sites).
// ---------------------------------------------------------------------------
__global__ __launch_bounds__(256) void gemm_bf16(
    const float* __restrict__ A, int lda,
    const float* __restrict__ B, int ldb,
    float* __restrict__ C, int ldc,
    int K, int Ncols, int relu, int bt, int normA,
    const float* __restrict__ dinv)
{
    __shared__ alignas(16) unsigned short As[128 * 40]; // row stride 40 shorts (80B, padded)
    __shared__ alignas(16) unsigned short Bs[128 * 40]; // stored transposed: Bs[n][k]

    const int tid   = threadIdx.x;
    const int mBase = blockIdx.y * 128;
    const int cBase = blockIdx.x * 128;
    const int lane = tid & 31, wid = tid >> 5;
    const int wm = wid & 1, wn = wid >> 1;
    const int laneHalf = lane >> 4, laneMod = lane & 15;

    v8f acc[4][2];
    #pragma unroll
    for (int mt = 0; mt < 4; ++mt)
        #pragma unroll
        for (int nt = 0; nt < 2; ++nt)
            #pragma unroll
            for (int i = 0; i < 8; ++i) acc[mt][nt][i] = 0.0f;

    for (int k0 = 0; k0 < K; k0 += 32) {
        // ---- stage A tile (128x32), fp32 -> bf16, optional fused sym-norm
        #pragma unroll
        for (int it = 0; it < 4; ++it) {
            int idx = tid + it * 256;
            int row = idx >> 3, c4 = idx & 7;
            int gr = mBase + row, gk = k0 + c4 * 4;
            f4v v = *(const f4v*)(A + (size_t)gr * lda + gk);
            if (normA) {
                float dr = dinv[gr];
                #pragma unroll
                for (int e = 0; e < 4; ++e) {
                    int col = gk + e;
                    float x = v[e] + ((gr == col) ? 1.0f : 0.0f);
                    v[e] = x * dr * dinv[col];
                }
            }
            u2v p; p[0] = pk2(v[0], v[1]); p[1] = pk2(v[2], v[3]);
            *(u2v*)(&As[row * 40 + c4 * 4]) = p;
        }
        // ---- stage B tile transposed into Bs[n][k]
        if (bt) {  // B is [Ncols x K] row-major: direct row-major copy
            #pragma unroll
            for (int it = 0; it < 4; ++it) {
                int idx = tid + it * 256;
                int row = idx >> 3, c4 = idx & 7;
                int n = cBase + row, gk = k0 + c4 * 4;
                f4v v = {0.f, 0.f, 0.f, 0.f};
                if (n < Ncols) v = *(const f4v*)(B + (size_t)n * ldb + gk);
                u2v p; p[0] = pk2(v[0], v[1]); p[1] = pk2(v[2], v[3]);
                *(u2v*)(&Bs[row * 40 + c4 * 4]) = p;
            }
        } else {   // B is [K x Ncols] row-major: scalar transpose into LDS
            #pragma unroll
            for (int it = 0; it < 16; ++it) {
                int idx = tid + it * 256;
                int c = idx & 127, kk = idx >> 7;
                int col = cBase + c;
                float val = (col < Ncols) ? B[(size_t)(k0 + kk) * ldb + col] : 0.0f;
                Bs[c * 40 + kk] = f2bf(val);
            }
        }
        __syncthreads();

        // ---- fragments (A: lanes0-15 K{kb..kb+7,kb+16..kb+23}, kb=8*half;
        //                 B: same K order as A, from transposed store)
        FragBF fa[4], fb[2];
        #pragma unroll
        for (int mt = 0; mt < 4; ++mt) {
            int r = wm * 64 + mt * 16 + laneMod;
            int kb = laneHalf * 8;
            fa[mt].q[0] = *(const u4v*)(&As[r * 40 + kb]);
            fa[mt].q[1] = *(const u4v*)(&As[r * 40 + kb + 16]);
        }
        #pragma unroll
        for (int nt = 0; nt < 2; ++nt) {
            int cc = wn * 32 + nt * 16 + laneMod;
            int kb = laneHalf * 8;
            fb[nt].q[0] = *(const u4v*)(&Bs[cc * 40 + kb]);
            fb[nt].q[1] = *(const u4v*)(&Bs[cc * 40 + kb + 16]);
        }
        #pragma unroll
        for (int mt = 0; mt < 4; ++mt)
            #pragma unroll
            for (int nt = 0; nt < 2; ++nt)
                acc[mt][nt] = __builtin_amdgcn_wmma_f32_16x16x32_bf16(
                    false, fa[mt].v, false, fb[nt].v, (short)0, acc[mt][nt],
                    false, false);
        __syncthreads();
    }

    // ---- epilogue: D layout (VGPR v, lane l): M = v + 8*(l>=16), N = l&15
    #pragma unroll
    for (int mt = 0; mt < 4; ++mt) {
        #pragma unroll
        for (int nt = 0; nt < 2; ++nt) {
            int col = cBase + wn * 32 + nt * 16 + laneMod;
            if (col < Ncols) {
                #pragma unroll
                for (int v = 0; v < 8; ++v) {
                    int row = mBase + wm * 64 + mt * 16 + v + 8 * laneHalf;
                    float x = acc[mt][nt][v];
                    if (relu) x = fmaxf(x, 0.0f);
                    C[(size_t)row * ldc + col] = x;
                }
            }
        }
    }
}

// ---------------------------------------------------------------------------
// Selection support kernels (exact radix-select in score space; integer
// atomics only -> deterministic).
// state: [0]=num_exist [1]=kA [2]=kB [3]=prefA [4]=prefB
//        [5]=thr_del bits [6]=thr_add bits [7]=delEn [8]=addEn
// ---------------------------------------------------------------------------
__global__ void zero_stuff(unsigned* st, unsigned* hA, unsigned* hB) {
    int t = blockIdx.x * blockDim.x + threadIdx.x;
    if (t < 2048) { hA[t] = 0u; hB[t] = 0u; }
    if (t < 16) st[t] = 0u;
}

__global__ __launch_bounds__(256) void count_exist(const float* __restrict__ orig,
                                                   unsigned* st) {
    __shared__ unsigned red[256];
    int i = blockIdx.x;
    unsigned cnt = 0;
    for (int j = i + 1 + threadIdx.x; j < Nn; j += 256)
        cnt += (orig[(size_t)i * Nn + j] > 0.0f) ? 1u : 0u;
    red[threadIdx.x] = cnt; __syncthreads();
    for (int s = 128; s > 0; s >>= 1) {
        if (threadIdx.x < s) red[threadIdx.x] += red[threadIdx.x + s];
        __syncthreads();
    }
    if (threadIdx.x == 0 && red[0]) atomicAdd(&st[0], red[0]);
}

__global__ void init_k(unsigned* st) {
    if (threadIdx.x == 0 && blockIdx.x == 0) {
        unsigned ne = st[0];
        int num_del = (int)floorf((float)ne * 0.1f);     // REMOVE_RATIO
        int n_add   = (int)floorf((float)ne * 0.1f);     // ADD_RATIO
        unsigned long long P = ((unsigned long long)Nn * (unsigned long long)(Nn - 1)) / 2ull;
        unsigned cnt_non = (unsigned)(P - (unsigned long long)ne);
        st[1] = (unsigned)(num_del < 1 ? 1 : num_del);   // k-th smallest existing
        unsigned na = (unsigned)(n_add < 1 ? 1 : n_add);
        st[2] = cnt_non - na + 1u;                        // ascending index of n_add-th largest
        st[3] = 0u; st[4] = 0u;
        st[7] = (num_del >= 1) ? 1u : 0u;
        st[8] = (n_add  >= 1) ? 1u : 0u;
    }
}

__global__ __launch_bounds__(256) void hist_pass(
    const float* __restrict__ logits, const float* __restrict__ orig,
    const unsigned* __restrict__ st, unsigned* hA, unsigned* hB,
    int shift, unsigned mask, int bins)
{
    int cx = blockIdx.x * 128, ry = blockIdx.y * 128;
    if (cx + 127 <= ry) return;   // block entirely at/below diagonal
    __shared__ unsigned sA[2048], sB[2048];
    for (int b = threadIdx.x; b < 2048; b += 256) { sA[b] = 0u; sB[b] = 0u; }
    __syncthreads();
    unsigned pA = st[3], pB = st[4];
    for (int it = 0; it < 64; ++it) {
        int idx = it * 256 + threadIdx.x;
        int r = ry + (idx >> 7), c = cx + (idx & 127);
        if (c > r) {
            size_t o = (size_t)r * Nn + c;
            float s = logits[o];
            bool e = orig[o] > 0.0f;
            unsigned key = ordkey(s);
            if ((key & mask) == (e ? pA : pB)) {
                unsigned b = (key >> shift) & (unsigned)(bins - 1);
                atomicAdd(e ? &sA[b] : &sB[b], 1u);
            }
        }
    }
    __syncthreads();
    for (int b = threadIdx.x; b < bins; b += 256) {
        if (sA[b]) atomicAdd(&hA[b], sA[b]);
        if (sB[b]) atomicAdd(&hB[b], sB[b]);
    }
}

__global__ void scan_pass(unsigned* st, unsigned* hA, unsigned* hB,
                          int shift, int bins, int last) {
    if (threadIdx.x != 0 || blockIdx.x != 0) return;
    {   // selection A (existing edges, k-th smallest)
        unsigned k = st[1], cum = 0; int found = -1;
        for (int b = 0; b < 2048; ++b) {
            unsigned c = (b < bins) ? hA[b] : 0u;
            if (found < 0 && b < bins) {
                if (cum + c >= k) { found = b; k -= cum; } else cum += c;
            }
            hA[b] = 0u;  // re-zero for next pass
        }
        if (found < 0) { found = bins - 1; k = 1; }
        st[1] = k; st[3] |= ((unsigned)found) << shift;
        if (last) st[5] = st[3];
    }
    {   // selection B (non-existing edges)
        unsigned k = st[2], cum = 0; int found = -1;
        for (int b = 0; b < 2048; ++b) {
            unsigned c = (b < bins) ? hB[b] : 0u;
            if (found < 0 && b < bins) {
                if (cum + c >= k) { found = b; k -= cum; } else cum += c;
            }
            hB[b] = 0u;
        }
        if (found < 0) { found = bins - 1; k = 1; }
        st[2] = k; st[4] |= ((unsigned)found) << shift;
        if (last) st[6] = st[4];
    }
}

// Build symmetric adj_new; both halves written coalesced via LDS transpose.
__global__ __launch_bounds__(256) void build_adj(
    const float* __restrict__ logits, const float* __restrict__ orig,
    const unsigned* __restrict__ st, float* __restrict__ out)
{
    int bx = blockIdx.x, by = blockIdx.y;
    if (bx < by) return;                   // only upper-or-diagonal 64x64 tiles
    int rbase = by * 64, cbase = bx * 64;
    __shared__ float tile[64][65];
    float thrA = ordinv(st[5]), thrB = ordinv(st[6]);
    bool delEn = st[7] != 0u, addEn = st[8] != 0u;
    for (int it = 0; it < 16; ++it) {
        int idx = it * 256 + threadIdx.x;
        int a = idx >> 6, b = idx & 63;
        int r = rbase + a, c = cbase + b;
        float val = 0.0f;
        if (c > r) {
            size_t o = (size_t)r * Nn + c;
            float s = logits[o], ev = orig[o];
            bool e = ev > 0.0f;
            bool del = delEn && e && (s < thrA);
            bool add = addEn && !e && (s >= thrB);
            val = add ? 1.0f : (del ? 0.0f : ev);
        }
        if (c >= r) out[(size_t)r * Nn + c] = val;   // upper triangle + diag(0)
        tile[a][b] = val;
    }
    __syncthreads();
    for (int it = 0; it < 16; ++it) {                // mirrored (strict lower) tile
        int idx = it * 256 + threadIdx.x;
        int aa = idx & 63, bb = idx >> 6;
        int rr = cbase + bb, cc = rbase + aa;
        if (rr > cc) out[(size_t)rr * Nn + cc] = tile[aa][bb];
    }
}

// dinv[i] = rsqrt(max(rowsum(adj_new)+1, 1e-12))   (+1 = self loop)
__global__ __launch_bounds__(256) void row_norm(const float* __restrict__ a,
                                                float* __restrict__ dinv) {
    __shared__ float red[256];
    int i = blockIdx.x; float s = 0.0f;
    for (int j = threadIdx.x; j < Nn; j += 256) s += a[(size_t)i * Nn + j];
    red[threadIdx.x] = s; __syncthreads();
    for (int t = 128; t > 0; t >>= 1) {
        if (threadIdx.x < t) red[threadIdx.x] += red[threadIdx.x + t];
        __syncthreads();
    }
    if (threadIdx.x == 0) dinv[i] = rsqrtf(fmaxf(red[0] + 1.0f, 1e-12f));
}

// ---------------------------------------------------------------------------
extern "C" void kernel_launch(void* const* d_in, const int* in_sizes, int n_in,
                              void* d_out, int out_size, void* d_ws, size_t ws_size,
                              hipStream_t stream)
{
    const float* feats   = (const float*)d_in[0];
    const float* adj     = (const float*)d_in[1];
    const float* adjorig = (const float*)d_in[2];
    const float* w_ep1   = (const float*)d_in[3];
    const float* w_ep2   = (const float*)d_in[4];
    const float* w_nc1   = (const float*)d_in[5];
    const float* w_nc2   = (const float*)d_in[6];

    float* outNC  = (float*)d_out;                       // [N,C]
    float* logits = outNC + (size_t)Nn * Cc;             // [N,N]
    float* adjNew = logits + (size_t)Nn * Nn;            // [N,N]

    float* XW1  = (float*)d_ws;                          // [N,H]
    float* H1   = XW1 + (size_t)Nn * Hh;                 // [N,H]
    float* HW2  = H1  + (size_t)Nn * Hh;                 // [N,E]
    float* Z    = HW2 + (size_t)Nn * Ee;                 // [N,E]
    float* dinv = Z   + (size_t)Nn * Ee;                 // [N]
    unsigned* hA = (unsigned*)(dinv + Nn);               // [2048]
    unsigned* hB = hA + 2048;                            // [2048]
    unsigned* st = hB + 2048;                            // [16]

    dim3 blk(256);

    // ---- ep_net: Z = relu(adj @ (relu(adj @ (X@W1)) @ W2)); logits = Z Z^T
    gemm_bf16<<<dim3(2, 64), blk, 0, stream>>>(feats, Ff, w_ep1, Hh, XW1, Hh, Ff, Hh, 0, 0, 0, nullptr);
    gemm_bf16<<<dim3(2, 64), blk, 0, stream>>>(adj,   Nn, XW1,  Hh, H1,  Hh, Nn, Hh, 1, 0, 0, nullptr);
    gemm_bf16<<<dim3(1, 64), blk, 0, stream>>>(H1,    Hh, w_ep2, Ee, HW2, Ee, Hh, Ee, 0, 0, 0, nullptr);
    gemm_bf16<<<dim3(1, 64), blk, 0, stream>>>(adj,   Nn, HW2,  Ee, Z,   Ee, Nn, Ee, 1, 0, 0, nullptr);
    gemm_bf16<<<dim3(64, 64), blk, 0, stream>>>(Z,    Ee, Z,    Ee, logits, Nn, Ee, Nn, 0, 1, 0, nullptr);

    // ---- exact threshold selection (radix select 11+11+10 bits)
    zero_stuff<<<8, blk, 0, stream>>>(st, hA, hB);
    count_exist<<<Nn, blk, 0, stream>>>(adjorig, st);
    init_k<<<1, 1, 0, stream>>>(st);
    hist_pass<<<dim3(64, 64), blk, 0, stream>>>(logits, adjorig, st, hA, hB, 21, 0x00000000u, 2048);
    scan_pass<<<1, 1, 0, stream>>>(st, hA, hB, 21, 2048, 0);
    hist_pass<<<dim3(64, 64), blk, 0, stream>>>(logits, adjorig, st, hA, hB, 10, 0xFFE00000u, 2048);
    scan_pass<<<1, 1, 0, stream>>>(st, hA, hB, 10, 2048, 0);
    hist_pass<<<dim3(64, 64), blk, 0, stream>>>(logits, adjorig, st, hA, hB, 0, 0xFFFFFC00u, 1024);
    scan_pass<<<1, 1, 0, stream>>>(st, hA, hB, 0, 1024, 1);

    // ---- rebuild adjacency + normalization vector
    build_adj<<<dim3(128, 128), blk, 0, stream>>>(logits, adjorig, st, adjNew);
    row_norm<<<Nn, blk, 0, stream>>>(adjNew, dinv);

    // ---- nc_net on sym_norm(adj_new), normalization fused into A loader
    gemm_bf16<<<dim3(2, 64), blk, 0, stream>>>(feats,  Ff, w_nc1, Hh, XW1,  Hh, Ff, Hh, 0, 0, 0, nullptr);
    gemm_bf16<<<dim3(2, 64), blk, 0, stream>>>(adjNew, Nn, XW1,  Hh, H1,   Hh, Nn, Hh, 1, 0, 1, dinv);
    gemm_bf16<<<dim3(1, 64), blk, 0, stream>>>(H1,     Hh, w_nc2, Cc, HW2,  Cc, Hh, Cc, 0, 0, 0, nullptr);
    gemm_bf16<<<dim3(1, 64), blk, 0, stream>>>(adjNew, Nn, HW2,  Cc, outNC, Cc, Nn, Cc, 0, 0, 1, dinv);

    (void)in_sizes; (void)n_in; (void)out_size; (void)ws_size;
}
